// ContextAttnDecoder_51015621542122
// MI455X (gfx1250) — matrix operand order repaired and checked by reading, not compiled, not run on hardware
//
#include <hip/hip_runtime.h>
#include <hip/hip_bf16.h>
#include <math.h>

#define B_    128
#define H_    512
#define E_    512
#define V_    50000
#define OOV_  50
#define VO_   (V_ + OOV_)     // 50050
#define TENC_ 128
#define TCTX_ 128
#define CH_   10              // softmax partial chunks per row (50000/10 = 5000)

typedef __attribute__((ext_vector_type(16))) __bf16 v16bf;
typedef __attribute__((ext_vector_type(8)))  float  v8f;

// Split fp32 into bf16 hi (truncated) + bf16 lo (residual). hi+lo carries ~16
// mantissa bits, so 3 bf16 WMMAs (hh + hl + lh) reproduce fp32 GEMM to ~2^-16.
__device__ inline void bsplit(float f, unsigned short& hi, unsigned short& lo) {
  unsigned u = __float_as_uint(f);
  unsigned short h = (unsigned short)(u >> 16);
  float hf = __uint_as_float(((unsigned)h) << 16);
  float l  = f - hf;
  hi = h;
  lo = (unsigned short)(__float_as_uint(l) >> 16);
}

// ---------------------------------------------------------------------------
// Generic WMMA GEMM core: OUT[b][n] = sum_k W[n][k] * X[b][k]  (+bias)
//   W: fp32 row-major (Nout x K), converted to bf16 hi/lo in registers.
//   X: pre-split bf16 hi/lo, stored TRANSPOSED as XT[k][b] (K x 128).
// One wave owns one 16-row v-tile and all 8 b-tiles (8 x v8f accumulators),
// so W bytes are read once per wave for the whole batch.
// ---------------------------------------------------------------------------
__device__ inline void gemm_accum(const float* __restrict__ W,
                                  const unsigned short* __restrict__ Xhi,
                                  const unsigned short* __restrict__ Xlo,
                                  int K, int row, int lane, v8f acc[8]) {
  const int kh = lane >> 4;                 // half-wave selects K sub-ranges
  const float* wr = W + (size_t)row * K;
  for (int k0 = 0; k0 < K; k0 += 32) {
    // A fragment (16x32 bf16 layout): lane holds K runs [k0+8h, +8) and
    // [k0+16+8h, +8)  -> four float4 loads from one row of W.
    float4 p0 = *(const float4*)(wr + k0 + 8 * kh);
    float4 p1 = *(const float4*)(wr + k0 + 8 * kh + 4);
    float4 p2 = *(const float4*)(wr + k0 + 16 + 8 * kh);
    float4 p3 = *(const float4*)(wr + k0 + 16 + 8 * kh + 4);
    float a[16];
    a[0]=p0.x; a[1]=p0.y; a[2]=p0.z; a[3]=p0.w;
    a[4]=p1.x; a[5]=p1.y; a[6]=p1.z; a[7]=p1.w;
    a[8]=p2.x; a[9]=p2.y; a[10]=p2.z; a[11]=p2.w;
    a[12]=p3.x; a[13]=p3.y; a[14]=p3.z; a[15]=p3.w;
    union { v16bf v; unsigned short u[16]; } ahi, alo;
#pragma unroll
    for (int e = 0; e < 16; ++e) bsplit(a[e], ahi.u[e], alo.u[e]);

    // B fragment: lane L holds column N = L%16, K runs [k0+8h, +8) and [k0+16+8h, +8).
    const unsigned short* bh = Xhi + (size_t)k0 * B_ + (lane & 15);
    const unsigned short* bl = Xlo + (size_t)k0 * B_ + (lane & 15);
#pragma unroll
    for (int bt = 0; bt < 8; ++bt) {
      union { v16bf v; unsigned short u[16]; } bhi, blo;
#pragma unroll
      for (int e = 0; e < 16; ++e) {
        int kk = (e < 8) ? (8 * kh + e) : (16 + 8 * kh + (e - 8));
        bhi.u[e] = bh[(size_t)kk * B_ + bt * 16];
        blo.u[e] = bl[(size_t)kk * B_ + bt * 16];
      }
      acc[bt] = __builtin_amdgcn_wmma_f32_16x16x32_bf16(false, ahi.v, false, bhi.v,
                                                        (short)0, acc[bt], false, false);
      acc[bt] = __builtin_amdgcn_wmma_f32_16x16x32_bf16(false, alo.v, false, bhi.v,
                                                        (short)0, acc[bt], false, false);
      acc[bt] = __builtin_amdgcn_wmma_f32_16x16x32_bf16(false, ahi.v, false, blo.v,
                                                        (short)0, acc[bt], false, false);
    }
  }
}

__global__ __launch_bounds__(128) void gemm_wmma_kernel(
    const float* __restrict__ W0, const float* __restrict__ bias0,
    const float* __restrict__ bias1,
    const unsigned short* __restrict__ X0hi, const unsigned short* __restrict__ X0lo, int K0,
    const float* __restrict__ W1,
    const unsigned short* __restrict__ X1hi, const unsigned short* __restrict__ X1lo, int K1,
    float* __restrict__ OUT, int Nout) {
  const int lane  = threadIdx.x & 31;
  const int wave  = threadIdx.x >> 5;
  const int vtile = blockIdx.x * 4 + wave;       // 4 waves / block
  const int Ntiles = Nout >> 4;
  if (vtile >= Ntiles) return;                   // wave-uniform: EXEC stays full

  v8f acc[8];
#pragma unroll
  for (int i = 0; i < 8; ++i) acc[i] = (v8f){0.f,0.f,0.f,0.f,0.f,0.f,0.f,0.f};

  const int row = vtile * 16 + (lane & 15);
  gemm_accum(W0, X0hi, X0lo, K0, row, lane, acc);
  if (W1) gemm_accum(W1, X1hi, X1lo, K1, row, lane, acc);

  // D layout: lane = N + 16*(M>=8), VGPR r = M&7
  const int Mbase = (lane >> 4) * 8;
  const int nb    = lane & 15;
  float bv[8];
#pragma unroll
  for (int r = 0; r < 8; ++r) {
    int n = vtile * 16 + Mbase + r;
    float bb = 0.f;
    if (bias0) bb += bias0[n];
    if (bias1) bb += bias1[n];
    bv[r] = bb;
  }
#pragma unroll
  for (int bt = 0; bt < 8; ++bt) {
    float* o = OUT + (size_t)(bt * 16 + nb) * Nout + vtile * 16 + Mbase;
#pragma unroll
    for (int r = 0; r < 8; ++r) o[r] = acc[bt][r] + bv[r];
  }
}

// ------------------------- prep: gather + transpose-split -------------------
__global__ void prep_kernel(const int* __restrict__ ids, const float* __restrict__ embedding,
                            const float* __restrict__ h0, float* __restrict__ emb,
                            unsigned short* eThi, unsigned short* eTlo,
                            unsigned short* hThi, unsigned short* hTlo) {
  int b = blockIdx.x, k = threadIdx.x;                 // 512 threads
  float e = embedding[(size_t)ids[b] * E_ + k];
  emb[b * E_ + k] = e;
  bsplit(e, eThi[k * B_ + b], eTlo[k * B_ + b]);
  float h = h0[b * H_ + k];
  bsplit(h, hThi[k * B_ + b], hTlo[k * B_ + b]);
}

// ------------------------- LSTM elementwise ---------------------------------
__global__ void lstm_kernel(const float* __restrict__ gates, const float* __restrict__ c0,
                            float* __restrict__ h1, float* __restrict__ c1,
                            unsigned short* hThi, unsigned short* hTlo,
                            float* __restrict__ out) {
  int b = blockIdx.x, k = threadIdx.x;                 // 512 threads
  const float* g = gates + (size_t)b * 4 * H_;
  float iv = g[k], fv = g[H_ + k], gv = g[2 * H_ + k], ov = g[3 * H_ + k];
  float si = 1.f / (1.f + expf(-iv));
  float sf = 1.f / (1.f + expf(-fv));
  float so = 1.f / (1.f + expf(-ov));
  float cc = sf * c0[b * H_ + k] + si * tanhf(gv);
  float hh = so * tanhf(cc);
  h1[b * H_ + k] = hh;
  c1[b * H_ + k] = cc;
  bsplit(hh, hThi[k * B_ + b], hTlo[k * B_ + b]);
  out[(size_t)B_ * VO_ + (size_t)b * H_ + k] = hh;                     // h1 output
  out[(size_t)B_ * VO_ + (size_t)B_ * H_ + (size_t)b * H_ + k] = cc;   // c1 output
}

// ------------------------- batched attention --------------------------------
__global__ void attn_kernel(const float* __restrict__ X, const float* __restrict__ dec,
                            int T, float* __restrict__ ctx, float* __restrict__ scores_out) {
  __shared__ float sc[TCTX_];
  __shared__ float red[8];
  int b = blockIdx.x, tid = threadIdx.x, lane = tid & 31, wave = tid >> 5;  // 256 thr
  const float* xb = X + (size_t)b * T * H_;
  const float* db = dec + (size_t)b * H_;
  for (int t = wave; t < T; t += 8) {
    const float* xr = xb + (size_t)t * H_;
    float s = 0.f;
    for (int k = lane; k < H_; k += 32) s += xr[k] * db[k];
#pragma unroll
    for (int off = 16; off; off >>= 1) s += __shfl_xor(s, off, 32);
    if (lane == 0) sc[t] = s;
  }
  __syncthreads();
  float v = (tid < T) ? sc[tid] : -3e38f;
  float m = v;
#pragma unroll
  for (int off = 16; off; off >>= 1) m = fmaxf(m, __shfl_xor(m, off, 32));
  if (lane == 0) red[wave] = m;
  __syncthreads();
  m = red[0];
  for (int w = 1; w < 8; ++w) m = fmaxf(m, red[w]);
  __syncthreads();
  float e = (tid < T) ? expf(v - m) : 0.f;
  float s = e;
#pragma unroll
  for (int off = 16; off; off >>= 1) s += __shfl_xor(s, off, 32);
  if (lane == 0) red[wave] = s;
  __syncthreads();
  s = 0.f;
  for (int w = 0; w < 8; ++w) s += red[w];
  if (tid < T) {
    float sv = e / s;
    sc[tid] = sv;
    if (scores_out) scores_out[(size_t)b * T + tid] = sv;
  }
  __syncthreads();
  for (int k = tid; k < H_; k += 256) {
    float a = 0.f;
    for (int t = 0; t < T; ++t) a += sc[t] * xb[(size_t)t * H_ + k];
    ctx[(size_t)b * H_ + k] = a;
  }
}

// ------------------------- p_gen ------------------------------------------
__global__ void pgen_kernel(const float* __restrict__ ctx, const float* __restrict__ ctxc,
                            const float* __restrict__ h1, const float* __restrict__ emb,
                            const float* __restrict__ gen_W, const float* __restrict__ gen_b,
                            const float* __restrict__ sigb, const int* __restrict__ cvar,
                            float* __restrict__ pgen, float* __restrict__ qgen) {
  __shared__ float red[8];
  __shared__ int   ired[8];
  int b = blockIdx.x, tid = threadIdx.x, lane = tid & 31, wave = tid >> 5;  // 256 thr
  float s = 0.f;
  for (int j = tid; j < 2048; j += 256) {
    float x;
    if (j < 512)       x = ctx[b * 512 + j];
    else if (j < 1024) x = ctxc[b * 512 + j - 512];
    else if (j < 1536) x = h1[b * 512 + j - 1024];
    else               x = emb[b * 512 + j - 1536];
    s += gen_W[j] * x;
  }
#pragma unroll
  for (int off = 16; off; off >>= 1) s += __shfl_xor(s, off, 32);
  if (lane == 0) red[wave] = s;
  int cnt = 0;
  for (int t = tid; t < TCTX_; t += 256) cnt += (cvar[b * TCTX_ + t] > 0) ? 1 : 0;
#pragma unroll
  for (int off = 16; off; off >>= 1) cnt += __shfl_xor(cnt, off, 32);
  if (lane == 0) ired[wave] = cnt;
  __syncthreads();
  if (tid == 0) {
    float ss = 0.f; int cc = 0;
    for (int w = 0; w < 8; ++w) { ss += red[w]; cc += ired[w]; }
    float pg = 1.f / (1.f + expf(-(ss + gen_b[0] + sigb[0])));
    if (cc == 0) { pgen[b] = 1.f; qgen[b] = 0.f; }
    else         { pgen[b] = pg;  qgen[b] = 1.f - pg; }
  }
}

// --------------- vocab softmax: online (max,sum) partials -------------------
__global__ void red1_kernel(const float* __restrict__ logits, float* Mp, float* Sp) {
  __shared__ float rm[8], rs[8];
  int b = blockIdx.x, c = blockIdx.y, tid = threadIdx.x, lane = tid & 31, wave = tid >> 5;
  const int chunk = V_ / CH_;
  const float* row = logits + (size_t)b * V_ + (size_t)c * chunk;
  float m = -3e38f, s = 0.f;
  for (int v = tid; v < chunk; v += 256) {
    float x = row[v];
    float nm = fmaxf(m, x);
    s = s * expf(m - nm) + expf(x - nm);
    m = nm;
  }
#pragma unroll
  for (int off = 16; off; off >>= 1) {
    float om = __shfl_xor(m, off, 32), os = __shfl_xor(s, off, 32);
    float nm = fmaxf(m, om);
    s = s * expf(m - nm) + os * expf(om - nm);
    m = nm;
  }
  if (lane == 0) { rm[wave] = m; rs[wave] = s; }
  __syncthreads();
  if (tid == 0) {
    m = rm[0]; s = rs[0];
    for (int w = 1; w < 8; ++w) {
      float nm = fmaxf(m, rm[w]);
      s = s * expf(m - nm) + rs[w] * expf(rm[w] - nm);
      m = nm;
    }
    Mp[b * CH_ + c] = m;
    Sp[b * CH_ + c] = s;
  }
}

__global__ void red2_kernel(const float* Mp, const float* Sp, float* M, float* S) {
  int b = threadIdx.x;                                   // 128 threads
  float m = Mp[b * CH_], s = Sp[b * CH_];
  for (int c = 1; c < CH_; ++c) {
    float om = Mp[b * CH_ + c], os = Sp[b * CH_ + c];
    float nm = fmaxf(m, om);
    s = s * expf(m - nm) + os * expf(om - nm);
    m = nm;
  }
  M[b] = m; S[b] = s;
}

// ------------------------- final log-prob -----------------------------------
__global__ void final_kernel(const float* __restrict__ logits, const float* M, const float* S,
                             const float* pgen, float* __restrict__ out) {
  int b = blockIdx.x;
  int v = blockIdx.y * 1024 + threadIdx.x;
  if (v >= VO_) return;
  float r;
  if (v < V_) {
    float p = pgen[b] * expf(logits[(size_t)b * V_ + v] - M[b]) / S[b];
    r = logf(fmaxf(p, 1e-10f));
  } else {
    r = -23.025850929940457f;                            // log(1e-10)
  }
  out[(size_t)b * VO_ + v] = r;
}

// ---- sparse copy-distribution patch (dedupe within each batch row) ---------
__global__ void upd_kernel(const int* __restrict__ cvar, const float* __restrict__ cscores,
                           const float* __restrict__ logits, const float* M, const float* S,
                           const float* pgen, const float* qgen, float* __restrict__ out) {
  __shared__ int   var[TCTX_];
  __shared__ float sc[TCTX_];
  int b = blockIdx.x, t = threadIdx.x;                   // 128 threads
  var[t] = cvar[b * TCTX_ + t];
  sc[t]  = cscores[b * TCTX_ + t];
  __syncthreads();
  int cv = var[t];
  bool first = true;
  float mass = 0.f;
  for (int u = 0; u < TCTX_; ++u) {
    if (var[u] == cv) { if (u < t) first = false; mass += sc[u]; }
  }
  if (first) {
    float pv = 0.f;
    if (cv < V_) pv = expf(logits[(size_t)b * V_ + cv] - M[b]) / S[b];
    float pr = pgen[b] * pv + qgen[b] * mass;
    out[(size_t)b * VO_ + cv] = logf(fmaxf(pr, 1e-10f));
  }
}

// ---------------------------------------------------------------------------
extern "C" void kernel_launch(void* const* d_in, const int* in_sizes, int n_in,
                              void* d_out, int out_size, void* d_ws, size_t ws_size,
                              hipStream_t stream) {
  const int*   ids     = (const int*)d_in[0];
  const float* h0      = (const float*)d_in[1];
  const float* c0      = (const float*)d_in[2];
  const float* enc     = (const float*)d_in[3];
  const float* ctxo    = (const float*)d_in[4];
  const int*   cvar    = (const int*)d_in[5];
  const float* embW    = (const float*)d_in[6];
  const float* W_ih    = (const float*)d_in[7];
  const float* W_hh    = (const float*)d_in[8];
  const float* b_ih    = (const float*)d_in[9];
  const float* b_hh    = (const float*)d_in[10];
  const float* attn_W  = (const float*)d_in[11];
  const float* attn_b  = (const float*)d_in[12];
  const float* cattn_W = (const float*)d_in[13];
  const float* cattn_b = (const float*)d_in[14];
  const float* gen_W   = (const float*)d_in[15];
  const float* gen_b   = (const float*)d_in[16];
  const float* sigb    = (const float*)d_in[17];
  const float* out_W   = (const float*)d_in[18];
  const float* out_b   = (const float*)d_in[19];
  float* out = (float*)d_out;

  char* p = (char*)d_ws;
  auto alloc = [&](size_t bytes) -> char* {
    char* r = p;
    p += (bytes + 255) & ~(size_t)255;
    return r;
  };
  float*          emb    = (float*)alloc((size_t)B_ * E_ * 4);
  unsigned short* eThi   = (unsigned short*)alloc((size_t)E_ * B_ * 2);
  unsigned short* eTlo   = (unsigned short*)alloc((size_t)E_ * B_ * 2);
  unsigned short* hThi   = (unsigned short*)alloc((size_t)H_ * B_ * 2);
  unsigned short* hTlo   = (unsigned short*)alloc((size_t)H_ * B_ * 2);
  float*          gates  = (float*)alloc((size_t)B_ * 4 * H_ * 4);
  float*          h1     = (float*)alloc((size_t)B_ * H_ * 4);
  float*          c1     = (float*)alloc((size_t)B_ * H_ * 4);
  unsigned short* h1Thi  = (unsigned short*)alloc((size_t)H_ * B_ * 2);
  unsigned short* h1Tlo  = (unsigned short*)alloc((size_t)H_ * B_ * 2);
  float*          dec    = (float*)alloc((size_t)B_ * H_ * 4);
  float*          dec2   = (float*)alloc((size_t)B_ * H_ * 4);
  float*          ctx    = (float*)alloc((size_t)B_ * H_ * 4);
  float*          ctxc   = (float*)alloc((size_t)B_ * H_ * 4);
  float*          cscr   = (float*)alloc((size_t)B_ * TCTX_ * 4);
  float*          pgen   = (float*)alloc((size_t)B_ * 4);
  float*          qgen   = (float*)alloc((size_t)B_ * 4);
  float*          Mp     = (float*)alloc((size_t)B_ * CH_ * 4);
  float*          Sp     = (float*)alloc((size_t)B_ * CH_ * 4);
  float*          Mv     = (float*)alloc((size_t)B_ * 4);
  float*          Sv     = (float*)alloc((size_t)B_ * 4);
  float*          logits = (float*)alloc((size_t)B_ * V_ * 4);   // 25.6 MB, last

  // 1) gather embedding + transpose/split inputs for WMMA B operands
  prep_kernel<<<B_, 512, 0, stream>>>(ids, embW, h0, emb, eThi, eTlo, hThi, hTlo);

  // 2) LSTM gates: dual-W accumulate GEMM (emb@W_ih^T + h0@W_hh^T + biases)
  gemm_wmma_kernel<<<(2048 / 16) / 4, 128, 0, stream>>>(
      W_ih, b_ih, b_hh, eThi, eTlo, 512, W_hh, hThi, hTlo, 512, gates, 4 * H_);

  // 3) LSTM nonlinearity -> h1, c1 (+ d_out tail, + h1^T hi/lo)
  lstm_kernel<<<B_, 512, 0, stream>>>(gates, c0, h1, c1, h1Thi, h1Tlo, out);

  // 4) attention projections (WMMA)
  gemm_wmma_kernel<<<(512 / 16) / 4, 128, 0, stream>>>(
      attn_W, attn_b, nullptr, h1Thi, h1Tlo, 512, nullptr, nullptr, nullptr, 0, dec, H_);
  gemm_wmma_kernel<<<(512 / 16) / 4, 128, 0, stream>>>(
      cattn_W, cattn_b, nullptr, h1Thi, h1Tlo, 512, nullptr, nullptr, nullptr, 0, dec2, H_);

  // 5) vocab projection: dominant GEMM, 3125 v-tiles (50000 = 16*3125)
  gemm_wmma_kernel<<<(3125 + 3) / 4, 128, 0, stream>>>(
      out_W, out_b, nullptr, h1Thi, h1Tlo, 512, nullptr, nullptr, nullptr, 0, logits, V_);

  // 6) attention contexts (batched matvec + softmax + weighted sum)
  attn_kernel<<<B_, 256, 0, stream>>>(enc, dec, TENC_, ctx, nullptr);
  attn_kernel<<<B_, 256, 0, stream>>>(ctxo, dec2, TCTX_, ctxc, cscr);

  // 7) p_gen
  pgen_kernel<<<B_, 256, 0, stream>>>(ctx, ctxc, h1, emb, gen_W, gen_b, sigb,
                                      cvar, pgen, qgen);

  // 8) vocab softmax reductions + fused final log-prob + sparse copy patch
  red1_kernel<<<dim3(B_, CH_), 256, 0, stream>>>(logits, Mp, Sp);
  red2_kernel<<<1, B_, 0, stream>>>(Mp, Sp, Mv, Sv);
  final_kernel<<<dim3(B_, (VO_ + 1023) / 1024), 1024, 0, stream>>>(logits, Mv, Sv, pgen, out);
  upd_kernel<<<B_, TCTX_, 0, stream>>>(cvar, cscr, logits, Mv, Sv, pgen, qgen, out);
}
